// BinaryClassifier_65146063946297
// MI455X (gfx1250) — matrix operand order repaired and checked by reading, not compiled, not run on hardware
//
#include <hip/hip_runtime.h>
#include <math.h>

// ---------------------------------------------------------------------------
// MI455X (gfx1250) implementation of the proposal-network forward pass.
// Large GEMMs: v_wmma_f32_16x16x32_bf16 with double-buffered LDS and
// GLOBAL_LOAD_ASYNC_TO_LDS_B128 staging of the A tiles (ASYNCcnt-tracked).
// ---------------------------------------------------------------------------

typedef __attribute__((ext_vector_type(16))) __bf16 v16bf;
typedef __attribute__((ext_vector_type(8)))  float  v8f;

#define LDK  40  // padded LDS row for bf16 tiles (80 B, 16B-aligned)
#define LDKF 40  // padded LDS row for fp32 tiles (160 B, 16B-aligned)

// --- async staging helpers (CDNA5 ISA 10.x / 15.18: ASYNCcnt-tracked) -------
// Copies 64 B (four b128) from global to LDS; INST_OFFSET advances BOTH the
// LDS and global address, so one LDS-addr VGPR + one 64-bit vaddr suffice.
__device__ __forceinline__ void async_copy_64B(uint32_t lds_off, const float* g) {
  asm volatile(
      "global_load_async_to_lds_b128 %0, %1, off\n\t"
      "global_load_async_to_lds_b128 %0, %1, off offset:16\n\t"
      "global_load_async_to_lds_b128 %0, %1, off offset:32\n\t"
      "global_load_async_to_lds_b128 %0, %1, off offset:48"
      :: "v"(lds_off), "v"(g) : "memory");
}
__device__ __forceinline__ void wait_async_zero() {
  asm volatile("s_wait_asynccnt 0x0" ::: "memory");
}
__device__ __forceinline__ uint32_t lds_offset_of(const void* p) {
  // LDS aperture: flat addr low 32 bits == LDS byte offset
  return (uint32_t)(uintptr_t)p;
}

// --- WMMA fragment loaders (layouts per CDNA5 ISA 7.12.2) -------------------
// A (16x32): lane l -> row m=l&15; elems 0..7 = K[h*8..], 8..15 = K[16+h*8..].
// Built from raw fp32 in LDS; v_cvt_pk_bf16_f32 co-executes with XDL WMMA.
__device__ __forceinline__ v16bf ld_a_frag_f32(const float* lAf, int rowBase,
                                               int m, int hlf) {
  const float* p = lAf + (size_t)(rowBase + m) * LDKF;
  float4 f0 = *(const float4*)(p + hlf * 8);
  float4 f1 = *(const float4*)(p + hlf * 8 + 4);
  float4 f2 = *(const float4*)(p + 16 + hlf * 8);
  float4 f3 = *(const float4*)(p + 16 + hlf * 8 + 4);
  v16bf r;
  r[0]  = (__bf16)f0.x; r[1]  = (__bf16)f0.y; r[2]  = (__bf16)f0.z; r[3]  = (__bf16)f0.w;
  r[4]  = (__bf16)f1.x; r[5]  = (__bf16)f1.y; r[6]  = (__bf16)f1.z; r[7]  = (__bf16)f1.w;
  r[8]  = (__bf16)f2.x; r[9]  = (__bf16)f2.y; r[10] = (__bf16)f2.z; r[11] = (__bf16)f2.w;
  r[12] = (__bf16)f3.x; r[13] = (__bf16)f3.y; r[14] = (__bf16)f3.z; r[15] = (__bf16)f3.w;
  return r;
}
// A from bf16 LDS (conv path).
__device__ __forceinline__ v16bf ld_a_frag(const __bf16* lA, int rowBase, int m, int hlf) {
  const __bf16* p = lA + (size_t)(rowBase + m) * LDK;
  v16bf r;
  ((uint4*)&r)[0] = *(const uint4*)(p + hlf * 8);
  ((uint4*)&r)[1] = *(const uint4*)(p + 16 + hlf * 8);
  return r;
}
// B (32x16), staged transposed [n][k]: lane l -> col n=l&15; elems 0..7 = K[h*8..], 8..15 = K[16+h*8..].
__device__ __forceinline__ v16bf ld_b_frag(const __bf16* lB, int nBase, int nn, int hlf) {
  const __bf16* p = lB + (size_t)(nBase + nn) * LDK;
  v16bf r;
  ((uint4*)&r)[0] = *(const uint4*)(p + hlf * 8);
  ((uint4*)&r)[1] = *(const uint4*)(p + 16 + hlf * 8);
  return r;
}

__device__ __forceinline__ float act_apply(float v, int ACT) {
  if (ACT == 1) {  // SELU
    const float sc = 1.0507009873554805f, al = 1.6732632423543772f;
    return v > 0.f ? sc * v : sc * al * (__expf(v) - 1.f);
  }
  if (ACT == 2) return v > 0.f ? v : 0.f;  // ReLU
  return v;
}

#define V8F_ZERO {0.f,0.f,0.f,0.f,0.f,0.f,0.f,0.f}

// ---------------------------------------------------------------------------
// bf16-WMMA GEMM, async-staged + double-buffered:
//   C = act(A[M,Kd] @ W[Kd,N] (+bias) (+extra))
// Block = 128 threads (4 waves), block tile 64x64, wave tile 32x32.
// A tiles: raw fp32 via GLOBAL_LOAD_ASYNC_TO_LDS_B128 (ping-pong buffers).
// B tiles: convert-to-bf16 in regs, transposed ds_store_b128.
// One workgroup barrier + one s_wait_asynccnt per K-step.
// M,N multiples of 64; Kd multiple of 32 (true at every call site).
// ---------------------------------------------------------------------------
template <int ACT, bool HAS_BIAS, bool HAS_EXTRA>
__global__ void __launch_bounds__(128) gemm_bf16_kernel(
    const float* __restrict__ A, const float* __restrict__ W,
    const float* __restrict__ bias, const float* __restrict__ extra,
    float* __restrict__ C, int M, int N, int Kd) {
  __shared__ float  lAf[2][64 * LDKF];   // 2 x 10 KB fp32 A tiles
  __shared__ __bf16 lB [2][64 * LDK];    // 2 x  5 KB bf16 B tiles (transposed)
  const int tid  = threadIdx.x;
  const int lane = tid & 31;
  const int wv   = tid >> 5;
  const int wm   = wv >> 1, wn = wv & 1;
  const int bm   = blockIdx.y * 64, bn = blockIdx.x * 64;
  const int m    = lane & 15, hlf = lane >> 4;

  v8f acc00 = V8F_ZERO, acc01 = V8F_ZERO, acc10 = V8F_ZERO, acc11 = V8F_ZERO;

  // A staging: thread -> (row = tid>>1, 16-float segment = (tid&1)*16)
  const int arow = tid >> 1;
  const int acol = (tid & 1) << 4;
  const uint32_t ldsA0 = lds_offset_of(&lAf[0][arow * LDKF + acol]);
  const uint32_t ldsA1 = lds_offset_of(&lAf[1][arow * LDKF + acol]);
  const float* agbase  = A + (size_t)(bm + arow) * Kd + acol;
  // B staging: thread -> (n = tid>>1, 16-k segment = (tid&1)*16)
  const int bnn = tid >> 1;
  const int bkk = (tid & 1) << 4;
  const float* wbase = W + bn + bnn;

  const int nK = Kd >> 5;

  // ---- prologue: stage tile 0 into buffer 0 ----
  async_copy_64B(ldsA0, agbase);
  {
    alignas(16) __bf16 tb[16];
    const float* p = wbase + (size_t)bkk * N;
#pragma unroll
    for (int i = 0; i < 16; i++) tb[i] = (__bf16)p[(size_t)i * N];
    __bf16* dst = &lB[0][bnn * LDK + bkk];
    ((uint4*)dst)[0] = ((uint4*)tb)[0];
    ((uint4*)dst)[1] = ((uint4*)tb)[1];
  }
  wait_async_zero();
  __syncthreads();

  for (int ki = 0; ki < nK; ki++) {
    const int cur = ki & 1;
    const bool more = (ki + 1) < nK;
    if (more) {  // stage next tile into the alternate buffer
      const int k1 = (ki + 1) << 5;
      async_copy_64B(cur ? ldsA0 : ldsA1, agbase + k1);
      alignas(16) __bf16 tb[16];
      const float* p = wbase + (size_t)(k1 + bkk) * N;
#pragma unroll
      for (int i = 0; i < 16; i++) tb[i] = (__bf16)p[(size_t)i * N];
      __bf16* dst = &lB[cur ^ 1][bnn * LDK + bkk];
      ((uint4*)dst)[0] = ((uint4*)tb)[0];
      ((uint4*)dst)[1] = ((uint4*)tb)[1];
    }
    v16bf a0 = ld_a_frag_f32(&lAf[cur][0], wm * 32,      m, hlf);
    v16bf a1 = ld_a_frag_f32(&lAf[cur][0], wm * 32 + 16, m, hlf);
    v16bf b0 = ld_b_frag(&lB[cur][0], wn * 32,      m, hlf);
    v16bf b1 = ld_b_frag(&lB[cur][0], wn * 32 + 16, m, hlf);
    acc00 = __builtin_amdgcn_wmma_f32_16x16x32_bf16(false, a0, false, b0, (short)0, acc00, false, false);
    acc01 = __builtin_amdgcn_wmma_f32_16x16x32_bf16(false, a0, false, b1, (short)0, acc01, false, false);
    acc10 = __builtin_amdgcn_wmma_f32_16x16x32_bf16(false, a1, false, b0, (short)0, acc10, false, false);
    acc11 = __builtin_amdgcn_wmma_f32_16x16x32_bf16(false, a1, false, b1, (short)0, acc11, false, false);
    if (more) {
      wait_async_zero();   // next A tile landed in LDS
      __syncthreads();     // next B tile visible; cur buffer reads done
    }
  }

  // Epilogue: C/D layout -> VGPR r holds row (r + 8*h), col = lane&15.
  const int rowBase = bm + wm * 32;
  const int colBase = bn + wn * 32;
#pragma unroll
  for (int im = 0; im < 2; im++) {
#pragma unroll
    for (int in = 0; in < 2; in++) {
      v8f a = (im == 0) ? (in == 0 ? acc00 : acc01) : (in == 0 ? acc10 : acc11);
      int col = colBase + in * 16 + m;
      float bv = HAS_BIAS ? bias[col] : 0.f;
#pragma unroll
      for (int r = 0; r < 8; r++) {
        int row = rowBase + im * 16 + r + 8 * hlf;
        float v = a[r] + bv;
        if (HAS_EXTRA) v += extra[(size_t)row * N + col];
        v = act_apply(v, ACT);
        C[(size_t)row * N + col] = v;
      }
    }
  }
}

// ---------------------------------------------------------------------------
// Conv1d(k=3, 256->256, pad 1) as WMMA GEMM, logical K = 3*256 = 768.
// Synchronous staging (boundary rows need zero-fill, which the async DMA
// cannot provide). In/Out layout: [B*T, 256] row-major.
// ---------------------------------------------------------------------------
template <int ACT>
__global__ void __launch_bounds__(128) conv3_gemm_kernel(
    const float* __restrict__ In, const float* __restrict__ Wc,
    const float* __restrict__ bias, float* __restrict__ Out) {
  const int N = 256;
  __shared__ __bf16 lA[64 * LDK];
  __shared__ __bf16 lB[64 * LDK];
  const int tid  = threadIdx.x;
  const int lane = tid & 31;
  const int wv   = tid >> 5;
  const int wm   = wv >> 1, wn = wv & 1;
  const int bm   = blockIdx.y * 64, bn = blockIdx.x * 64;
  const int m    = lane & 15, hlf = lane >> 4;

  v8f acc00 = V8F_ZERO, acc01 = V8F_ZERO, acc10 = V8F_ZERO, acc11 = V8F_ZERO;

  const int arow = tid >> 1;
  const int acol = (tid & 1) << 4;
  __bf16* lAw = &lA[arow * LDK + acol];
  const int bnn = tid >> 1;
  const int bkk = (tid & 1) << 4;
  __bf16* lBw = &lB[bnn * LDK + bkk];

  const int g  = bm + arow;        // global row in [0, B*T)
  const int bb = g >> 12;          // T = 4096
  const int tt = g & 4095;

  for (int k0 = 0; k0 < 768; k0 += 32) {
    const int tap = k0 >> 8;          // one tap per 32-wide K tile
    const int cib = k0 & 255;
    const int ts  = tt + tap - 1;
    alignas(16) __bf16 ta[16];
    if ((unsigned)ts < 4096u) {
      const float* p = In + ((size_t)((bb << 12) + ts)) * 256 + cib + acol;
      if (k0 + 32 < 768) __builtin_prefetch(p + 32, 0, 1);  // global_prefetch
#pragma unroll
      for (int i = 0; i < 4; i++) {
        float4 f = ((const float4*)p)[i];
        ta[4 * i + 0] = (__bf16)f.x; ta[4 * i + 1] = (__bf16)f.y;
        ta[4 * i + 2] = (__bf16)f.z; ta[4 * i + 3] = (__bf16)f.w;
      }
    } else {
#pragma unroll
      for (int i = 0; i < 16; i++) ta[i] = (__bf16)0.f;
    }
    alignas(16) __bf16 tb[16];
    {
      // conv_w layout [O=256][I=256][3]; B[k][n] = w[n][ci][tap]
      const float* p = Wc + (size_t)(bn + bnn) * 768 + (size_t)(cib + bkk) * 3 + tap;
#pragma unroll
      for (int i = 0; i < 16; i++) tb[i] = (__bf16)p[i * 3];
    }
    ((uint4*)lAw)[0] = ((uint4*)ta)[0];
    ((uint4*)lAw)[1] = ((uint4*)ta)[1];
    ((uint4*)lBw)[0] = ((uint4*)tb)[0];
    ((uint4*)lBw)[1] = ((uint4*)tb)[1];
    __syncthreads();

    v16bf a0 = ld_a_frag(lA, wm * 32,      m, hlf);
    v16bf a1 = ld_a_frag(lA, wm * 32 + 16, m, hlf);
    v16bf b0 = ld_b_frag(lB, wn * 32,      m, hlf);
    v16bf b1 = ld_b_frag(lB, wn * 32 + 16, m, hlf);
    acc00 = __builtin_amdgcn_wmma_f32_16x16x32_bf16(false, a0, false, b0, (short)0, acc00, false, false);
    acc01 = __builtin_amdgcn_wmma_f32_16x16x32_bf16(false, a0, false, b1, (short)0, acc01, false, false);
    acc10 = __builtin_amdgcn_wmma_f32_16x16x32_bf16(false, a1, false, b0, (short)0, acc10, false, false);
    acc11 = __builtin_amdgcn_wmma_f32_16x16x32_bf16(false, a1, false, b1, (short)0, acc11, false, false);
    __syncthreads();
  }

  const int rowBase = bm + wm * 32;
  const int colBase = bn + wn * 32;
#pragma unroll
  for (int im = 0; im < 2; im++) {
#pragma unroll
    for (int in = 0; in < 2; in++) {
      v8f a = (im == 0) ? (in == 0 ? acc00 : acc01) : (in == 0 ? acc10 : acc11);
      int col = colBase + in * 16 + m;
      float bv = bias[col];
#pragma unroll
      for (int r = 0; r < 8; r++) {
        int row = rowBase + im * 16 + r + 8 * hlf;
        float v = act_apply(a[r] + bv, ACT);
        Out[(size_t)row * N + col] = v;
      }
    }
  }
}

// --- actness[b,t] = sigmoid(enc_out[b,t,:] . W_sc[:,0] + b_sc[0]) * mask ----
__global__ void __launch_bounds__(256) actness_kernel(
    const float* __restrict__ enc, const float* __restrict__ Wsc,
    const float* __restrict__ bsc, const float* __restrict__ fmask,
    float* __restrict__ act) {
  int row  = blockIdx.x * 8 + (threadIdx.x >> 5);
  int lane = threadIdx.x & 31;
  const float* p = enc + (size_t)row * 256;
  float s = 0.f;
  for (int i = lane; i < 256; i += 32) s += p[i] * Wsc[i * 3];
#pragma unroll
  for (int off = 16; off > 0; off >>= 1) s += __shfl_xor(s, off);
  if (lane == 0) {
    float v = 1.f / (1.f + __expf(-(s + bsc[0])));
    act[row] = v * fmask[row];
  }
}

// --- per-batch iterative top-64 (first-occurrence tie-break, like top_k) ----
__global__ void __launch_bounds__(256) topk_kernel(
    const float* __restrict__ act, float* __restrict__ topv, float* __restrict__ topi) {
  __shared__ float buf[4096];
  __shared__ float rv[256];
  __shared__ int   ri[256];
  const int b = blockIdx.x;
  for (int i = threadIdx.x; i < 4096; i += 256) buf[i] = act[b * 4096 + i];
  __syncthreads();
  for (int it = 0; it < 64; it++) {
    float best = -1e30f;
    int   bi   = 0x7fffffff;
    for (int i = threadIdx.x; i < 4096; i += 256) {
      float v = buf[i];
      if (v > best || (v == best && i < bi)) { best = v; bi = i; }
    }
    rv[threadIdx.x] = best; ri[threadIdx.x] = bi;
    __syncthreads();
    for (int s2 = 128; s2 > 0; s2 >>= 1) {
      if (threadIdx.x < s2) {
        float ov = rv[threadIdx.x + s2]; int oi = ri[threadIdx.x + s2];
        if (ov > rv[threadIdx.x] || (ov == rv[threadIdx.x] && oi < ri[threadIdx.x])) {
          rv[threadIdx.x] = ov; ri[threadIdx.x] = oi;
        }
      }
      __syncthreads();
    }
    if (threadIdx.x == 0) {
      topv[b * 64 + it] = rv[0];
      topi[b * 64 + it] = (float)ri[0];
      buf[ri[0]] = -1e30f;
    }
    __syncthreads();
  }
}

// --- ROI build + pos-emb + 3-region ROI pooling gather ----------------------
__global__ void __launch_bounds__(128) roi_kernel(
    const float* __restrict__ enc, const float* __restrict__ topv,
    const float* __restrict__ topi, float* __restrict__ xpool,
    float* __restrict__ posemb, float* __restrict__ mask,
    float* __restrict__ out_rois, float* __restrict__ out_act) {
  const int gk = blockIdx.x;   // b*K + k, K = 256
  const int b  = gk >> 8;
  const int k  = gk & 255;
  const int kc = k >> 2;       // NS = 4
  const int si = k & 3;
  const float scl[4] = {16.f, 64.f, 128.f, 256.f};
  const float Tf = 4096.f;
  float tv = topv[b * 64 + kc];
  float c  = topi[b * 64 + kc];
  float hw = scl[si] * 0.5f;
  float s  = fminf(fmaxf(c - hw, 0.f), Tf - 1.f);
  float e  = fminf(fmaxf(c + hw, 1.f), Tf);
  e = fmaxf(e, s + 1.f);
  float L  = e - s;
  float ss = fminf(fmaxf(s - L * 0.25f, 0.f), Tf - 1.f);
  float se = s + L * 0.25f;
  float es = e - L * 0.25f;
  float ee = fminf(fmaxf(e + L * 0.25f, 1.f), Tf);
  if (threadIdx.x == 0) {
    out_rois[gk * 2 + 0] = s;
    out_rois[gk * 2 + 1] = e;
    out_act[gk]          = tv;
    mask[gk]             = (tv > 0.5f) ? 1.f : 0.f;
  }
  // sinusoidal pos-emb: [c2*128 + {sin:j, cos:64+j}], d4=64
  {
    int c2 = threadIdx.x >> 6;       // 0..1
    int j  = threadIdx.x & 63;       // 0..63
    float rel  = (c2 == 0) ? (s / Tf) : (e / Tf);
    float freq = __expf(-9.210340371976184f * (float)j / 64.f);  // ln(1e4)
    float ang  = rel * freq;
    posemb[(size_t)gk * 256 + c2 * 128 + j]      = __sinf(ang);
    posemb[(size_t)gk * 256 + c2 * 128 + 64 + j] = __cosf(ang);
  }
  // pooling: 3 regions x P=4 samples x 256 channels
  const float r0s[3] = {ss, s, es};
  const float r0e[3] = {se, e, ee};
  for (int seg = 0; seg < 12; seg++) {
    int rg = seg >> 2, p = seg & 3;
    float frac = ((float)p + 0.5f) * 0.25f;
    float t = floorf(r0s[rg] + (r0e[rg] - r0s[rg]) * frac);
    int ti = (int)fminf(fmaxf(t, 0.f), Tf - 1.f);
    const float* src = enc + ((size_t)(b << 12) + ti) * 256;
    float* dst = xpool + (size_t)gk * 3072 + seg * 256;
    for (int d = threadIdx.x; d < 256; d += 128) dst[d] = src[d];
  }
}

// --- attention core per (b,h): streaming softmax, K staged in LDS -----------
__global__ void __launch_bounds__(256) attn_kernel(
    const float* __restrict__ Q, const float* __restrict__ Km,
    const float* __restrict__ V, const float* __restrict__ mask,
    float* __restrict__ ctx) {
  __shared__ float lK[256 * 32];
  __shared__ float lbias[256];
  const int bh = blockIdx.x;
  const int b  = bh >> 3, h = bh & 7;
  const int q  = threadIdx.x;
  {
    const float* kp = Km + ((size_t)(b * 256 + q)) * 256 + h * 32;
#pragma unroll
    for (int j = 0; j < 32; j++) lK[q * 32 + j] = kp[j];
    lbias[q] = (1.f - mask[b * 256 + q]) * -1e9f;
  }
  __syncthreads();
  float qv[32];
  const float* qp = Q + ((size_t)(b * 256 + q)) * 256 + h * 32;
#pragma unroll
  for (int j = 0; j < 32; j++) qv[j] = qp[j] * 0.17677669529663687f;  // 1/sqrt(32)
  float mx = -1e30f;
  for (int kk = 0; kk < 256; kk++) {
    float sd = 0.f;
#pragma unroll
    for (int j = 0; j < 32; j++) sd += qv[j] * lK[kk * 32 + j];
    mx = fmaxf(mx, sd + lbias[kk]);
  }
  float den = 0.f;
  float o[32];
#pragma unroll
  for (int j = 0; j < 32; j++) o[j] = 0.f;
  const float* vb = V + ((size_t)(b * 256)) * 256 + h * 32;
  for (int kk = 0; kk < 256; kk++) {
    float sd = 0.f;
#pragma unroll
    for (int j = 0; j < 32; j++) sd += qv[j] * lK[kk * 32 + j];
    float p = __expf(sd + lbias[kk] - mx);
    den += p;
    const float* vp = vb + (size_t)kk * 256;
#pragma unroll
    for (int j = 0; j < 32; j++) o[j] += p * vp[j];
  }
  float inv = 1.f / den;
  float* op = ctx + ((size_t)(b * 256 + q)) * 256 + h * 32;
#pragma unroll
  for (int j = 0; j < 32; j++) op[j] = o[j] * inv;
}

// --- fused residual add + LayerNorm over 256 channels -----------------------
__global__ void __launch_bounds__(256) ln_add_kernel(
    const float* __restrict__ X, const float* __restrict__ Y,
    const float* __restrict__ g, const float* __restrict__ bta,
    float* __restrict__ out) {
  __shared__ float red[256];
  const int row = blockIdx.x, tid = threadIdx.x;
  float v = X[(size_t)row * 256 + tid] + Y[(size_t)row * 256 + tid];
  red[tid] = v;
  __syncthreads();
  for (int s2 = 128; s2 > 0; s2 >>= 1) {
    if (tid < s2) red[tid] += red[tid + s2];
    __syncthreads();
  }
  float mean = red[0] * (1.f / 256.f);
  __syncthreads();
  float d = v - mean;
  red[tid] = d * d;
  __syncthreads();
  for (int s2 = 128; s2 > 0; s2 >>= 1) {
    if (tid < s2) red[tid] += red[tid + s2];
    __syncthreads();
  }
  float var = red[0] * (1.f / 256.f);
  out[(size_t)row * 256 + tid] = d * rsqrtf(var + 1e-6f) * g[tid] + bta[tid];
}

// --- classifier: softmax(x2 @ W_cls + b_cls) over 2 classes ------------------
__global__ void __launch_bounds__(256) cls_kernel(
    const float* __restrict__ x2, const float* __restrict__ W,
    const float* __restrict__ bb, float* __restrict__ out) {
  int row = blockIdx.x * 256 + threadIdx.x;
  if (row >= 2048) return;
  float l0 = bb[0], l1 = bb[1];
  const float* p = x2 + (size_t)row * 256;
  for (int i = 0; i < 256; i++) {
    float v = p[i];
    l0 += v * W[i * 2];
    l1 += v * W[i * 2 + 1];
  }
  float mmx = fmaxf(l0, l1);
  float e0 = __expf(l0 - mmx), e1 = __expf(l1 - mmx);
  float inv = 1.f / (e0 + e1);
  out[row * 2 + 0] = e0 * inv;
  out[row * 2 + 1] = e1 * inv;
}

// ---------------------------------------------------------------------------
extern "C" void kernel_launch(void* const* d_in, const int* in_sizes, int n_in,
                              void* d_out, int out_size, void* d_ws, size_t ws_size,
                              hipStream_t stream) {
  (void)in_sizes; (void)n_in; (void)out_size; (void)ws_size;
  const float* feature = (const float*)d_in[0];
  const float* fmask   = (const float*)d_in[2];
  const float* W_red   = (const float*)d_in[3];
  const float* b_red   = (const float*)d_in[4];
  const float* conv1_w = (const float*)d_in[5];
  const float* conv1_b = (const float*)d_in[6];
  const float* conv2_w = (const float*)d_in[7];
  const float* conv2_b = (const float*)d_in[8];
  const float* W_sc    = (const float*)d_in[9];
  const float* b_sc    = (const float*)d_in[10];
  const float* W_pool  = (const float*)d_in[11];
  const float* b_pool  = (const float*)d_in[12];
  const float* Wq      = (const float*)d_in[13];
  const float* Wk      = (const float*)d_in[14];
  const float* Wv      = (const float*)d_in[15];
  const float* Wo      = (const float*)d_in[16];
  const float* ln1_g   = (const float*)d_in[17];
  const float* ln1_b   = (const float*)d_in[18];
  const float* W1      = (const float*)d_in[19];
  const float* b1      = (const float*)d_in[20];
  const float* W2      = (const float*)d_in[21];
  const float* b2      = (const float*)d_in[22];
  const float* ln2_g   = (const float*)d_in[23];
  const float* ln2_b   = (const float*)d_in[24];
  const float* W_cls   = (const float*)d_in[25];
  const float* b_cls   = (const float*)d_in[26];

  // Workspace layout (floats). Region reuse by stream order:
  //   [0,SZ)      ENC  (enc_input, needed through pooling)
  //   [SZ,2SZ)    conv1 out, later reused as XPOOL
  //   [2SZ,3SZ)   conv2 out (enc_output), later reused for ROI-stage buffers
  float* ws = (float*)d_ws;
  const size_t SZ = 8388608ull;   // 32768 * 256
  const size_t C5 = 524288ull;    // 2048 * 256
  float* ENC    = ws;
  float* H1     = ws + SZ;        // conv1 output
  float* XPOOL  = ws + SZ;        // reuse after conv2 consumed H1 (2048 x 3072)
  float* R2     = ws + 2 * SZ;
  float* ENCOUT = R2;             // conv2 output (consumed by actness)
  float* POSEMB = R2;             // reuse after actness
  float* X0     = R2 + 1 * C5;
  float* Qb     = R2 + 2 * C5;
  float* Kb     = R2 + 3 * C5;
  float* Vb     = R2 + 4 * C5;
  float* CTX    = R2 + 5 * C5;
  float* PROJ   = R2 + 6 * C5;
  float* X1     = R2 + 7 * C5;
  float* FFN1   = R2 + 8 * C5;    // 2048 x 1024 (4*C5)
  float* FFN2   = R2 + 12 * C5;
  float* X2     = R2 + 13 * C5;
  float* ACT    = ws + 3 * SZ;    // 32768
  float* TOPV   = ACT + 32768;    // 512
  float* TOPI   = TOPV + 512;     // 512 (indices stored exactly as floats)
  float* MASK   = TOPI + 512;     // 2048

  float* out        = (float*)d_out;
  float* out_rois   = out;          // [B,K,2] = 4096
  float* out_act    = out + 4096;   // [B,K]   = 2048
  float* out_scores = out + 6144;   // [B,K,2] = 4096

  const dim3 blk(128);
  // 1) enc_input = selu(feature @ W_red + b_red)           M=32768,N=256,K=1024
  gemm_bf16_kernel<1, true, false><<<dim3(4, 512), blk, 0, stream>>>(
      feature, W_red, b_red, nullptr, ENC, 32768, 256, 1024);
  // 2-3) two conv1d(k=3) + relu as WMMA GEMMs (K=768)
  conv3_gemm_kernel<2><<<dim3(4, 512), blk, 0, stream>>>(ENC, conv1_w, conv1_b, H1);
  conv3_gemm_kernel<2><<<dim3(4, 512), blk, 0, stream>>>(H1, conv2_w, conv2_b, ENCOUT);
  // 4) actness (only column 0 of W_sc feeds the outputs)
  actness_kernel<<<dim3(4096), dim3(256), 0, stream>>>(ENCOUT, W_sc, b_sc, fmask, ACT);
  // 5) per-batch top-64 peaks
  topk_kernel<<<dim3(8), dim3(256), 0, stream>>>(ACT, TOPV, TOPI);
  // 6) ROIs + pos-emb + pooling gather (writes rois_se, roi_actness)
  roi_kernel<<<dim3(2048), dim3(128), 0, stream>>>(
      ENC, TOPV, TOPI, XPOOL, POSEMB, MASK, out_rois, out_act);
  // 7) x0 = XPOOL @ W_pool + b_pool + pos_emb              M=2048,N=256,K=3072
  gemm_bf16_kernel<0, true, true><<<dim3(4, 32), blk, 0, stream>>>(
      XPOOL, W_pool, b_pool, POSEMB, X0, 2048, 256, 3072);
  // 8) Q,K,V projections (no bias)
  gemm_bf16_kernel<0, false, false><<<dim3(4, 32), blk, 0, stream>>>(
      X0, Wq, nullptr, nullptr, Qb, 2048, 256, 256);
  gemm_bf16_kernel<0, false, false><<<dim3(4, 32), blk, 0, stream>>>(
      X0, Wk, nullptr, nullptr, Kb, 2048, 256, 256);
  gemm_bf16_kernel<0, false, false><<<dim3(4, 32), blk, 0, stream>>>(
      X0, Wv, nullptr, nullptr, Vb, 2048, 256, 256);
  // 9) masked softmax attention per (b,h)
  attn_kernel<<<dim3(64), dim3(256), 0, stream>>>(Qb, Kb, Vb, MASK, CTX);
  // 10) output projection
  gemm_bf16_kernel<0, false, false><<<dim3(4, 32), blk, 0, stream>>>(
      CTX, Wo, nullptr, nullptr, PROJ, 2048, 256, 256);
  // 11) x1 = LN(x0 + proj)
  ln_add_kernel<<<dim3(2048), dim3(256), 0, stream>>>(X0, PROJ, ln1_g, ln1_b, X1);
  // 12-13) FFN
  gemm_bf16_kernel<2, true, false><<<dim3(16, 32), blk, 0, stream>>>(
      X1, W1, b1, nullptr, FFN1, 2048, 1024, 256);
  gemm_bf16_kernel<0, true, false><<<dim3(4, 32), blk, 0, stream>>>(
      FFN1, W2, b2, nullptr, FFN2, 2048, 256, 1024);
  // 14) x2 = LN(x1 + ffn)
  ln_add_kernel<<<dim3(2048), dim3(256), 0, stream>>>(X1, FFN2, ln2_g, ln2_b, X2);
  // 15) roi_scores = softmax(x2 @ W_cls + b_cls)
  cls_kernel<<<dim3(8), dim3(256), 0, stream>>>(X2, W_cls, b_cls, out_scores);
}
